// GNN_62886911148736
// MI455X (gfx1250) — matrix operand order repaired and checked by reading, not compiled, not run on hardware
//
#include <hip/hip_runtime.h>

// ---------------------------------------------------------------------------
// GNN (MeshGraphNets-style) forward for MI455X / gfx1250, wave32 + WMMA f16.
// ---------------------------------------------------------------------------

typedef __attribute__((ext_vector_type(16))) _Float16 v16h;
typedef __attribute__((ext_vector_type(8)))  _Float16 v8h;
typedef __attribute__((ext_vector_type(4)))  _Float16 v4h;
typedef __attribute__((ext_vector_type(8)))  float    v8f;

#define HIDDEN 128

#define WMMA_F16(a, b, c) \
  __builtin_amdgcn_wmma_f32_16x16x32_f16(false, (a), false, (b), (short)0, (c), false, false)

// ---------------------------------------------------------------------------
// Fragment loaders
// ---------------------------------------------------------------------------
// A fragment (16x32, M x K), 16-bit layout per CDNA5 ISA:
//   lane l (<16):  row M=l,    K = k0+{0..7}  then k0+{16..23}
//   lane l (>=16): row M=l-16, K = k0+{8..15} then k0+{24..31}
__device__ __forceinline__ v16h afrag(const _Float16* __restrict__ base, int lda,
                                      int row, int k0, int hiHalf) {
  const _Float16* p = base + (size_t)row * lda + k0 + hiHalf * 8;
  v8h lo = *(const v8h*)p;
  v8h hi = *(const v8h*)(p + 16);
  return __builtin_shufflevector(lo, hi, 0, 1, 2, 3, 4, 5, 6, 7,
                                 8, 9, 10, 11, 12, 13, 14, 15);
}

// B fragment from pre-fragmentized weights: [kt][nt][lane][elem0..15] contiguous.
__device__ __forceinline__ v16h bfrag(const _Float16* __restrict__ Wf,
                                      int kt, int nt, int ntiles, int lane) {
  const _Float16* p = Wf + (((size_t)(kt * ntiles + nt) * 32 + lane) * 16);
  __builtin_prefetch(p + 512, 0, 3);  // next n-tile's fragments (nt+1), near caches
  v8h lo = *(const v8h*)p;
  v8h hi = *(const v8h*)(p + 8);
  return __builtin_shufflevector(lo, hi, 0, 1, 2, 3, 4, 5, 6, 7,
                                 8, 9, 10, 11, 12, 13, 14, 15);
}

// Store C tiles (+bias, ReLU) into per-wave LDS activation staging, row-major f16.
// D layout: VGPR r, lane<16 -> (M=r, N=lane); lane>=16 -> (M=r+8, N=lane-16).
__device__ __forceinline__ void stash_relu(const v8f* accs,
                                           const float* __restrict__ bvec,
                                           _Float16* act, int lane) {
  const int colb = lane & 15;
  const int r0 = (lane & 16) ? 8 : 0;
#pragma unroll
  for (int nt = 0; nt < 8; ++nt) {
    const int col = nt * 16 + colb;
    const float bias = bvec[col];
#pragma unroll
    for (int r = 0; r < 8; ++r) {
      float v = accs[nt][r] + bias;
      v = fmaxf(v, 0.0f);
      act[(r0 + r) * HIDDEN + col] = (_Float16)v;
    }
  }
}

// ---------------------------------------------------------------------------
// Fused 3-layer MLP:  A[M x K0T*32] -> 128 -> 128 -> n2cols (ReLU between).
// All trip counts are compile-time: accumulators live entirely in VGPRs,
// kt-outer/nt-inner gives 8 WMMAs per A-fragment load.
//   MODE 0: dense A0 (lda0)
//   MODE 1: edge concat: kt 0..3 -> A0[idx0[e]], 4..7 -> A0[idx1[e]], 8..11 -> A1[e]
//   MODE 2: node concat: kt 0..3 -> A0[r] (h16), 4..7 -> A1[r] (agg16)
// ---------------------------------------------------------------------------
template <int MODE, int K0T, int N2T>
__global__ __launch_bounds__(256) void mlp3_kernel(
    const _Float16* __restrict__ A0, int lda0,
    const _Float16* __restrict__ A1, const int* __restrict__ idx0,
    const int* __restrict__ idx1, int M,
    const _Float16* __restrict__ W0, const float* __restrict__ b0,
    const _Float16* __restrict__ W1, const float* __restrict__ b1,
    const _Float16* __restrict__ W2, const float* __restrict__ b2,
    int n2cols, float* __restrict__ out, int ldo) {
  __shared__ __align__(16) _Float16 lds_act[8 * 16 * HIDDEN];  // 32 KB

  const int lane = threadIdx.x & 31;
  const int wave = threadIdx.x >> 5;
  const int rowBase = blockIdx.x * 128 + wave * 16;
  const int rl = lane & 15;
  const int hiHalf = (lane >> 4) & 1;
  _Float16* act = lds_act + wave * 16 * HIDDEN;

  v8f accs[8];
  auto zero_accs = [&]() {
#pragma unroll
    for (int nt = 0; nt < 8; ++nt) accs[nt] = v8f{};
  };
  // One A fragment feeds all 8 output tiles (operand reuse).
  auto accum = [&](v16h a, const _Float16* __restrict__ Wf, int kt) {
#pragma unroll
    for (int nt = 0; nt < 8; ++nt) {
      v16h b = bfrag(Wf, kt, nt, 8, lane);
      accs[nt] = WMMA_F16(a, b, accs[nt]);
    }
  };

  // ---------------- layer 0 (global/gathered A) ----------------
  zero_accs();
  if (MODE == 0) {
#pragma unroll
    for (int kt = 0; kt < K0T; ++kt)
      accum(afrag(A0, lda0, rowBase + rl, kt * 32, hiHalf), W0, kt);
  } else if (MODE == 1) {
    const int e = rowBase + rl;
    const int rs = idx0[e];
    const int rd = idx1[e];
#pragma unroll
    for (int kt = 0; kt < 4; ++kt)
      accum(afrag(A0, HIDDEN, rs, kt * 32, hiHalf), W0, kt);
#pragma unroll
    for (int kt = 0; kt < 4; ++kt)
      accum(afrag(A0, HIDDEN, rd, kt * 32, hiHalf), W0, kt + 4);
#pragma unroll
    for (int kt = 0; kt < 4; ++kt)
      accum(afrag(A1, HIDDEN, e, kt * 32, hiHalf), W0, kt + 8);
  } else {
    const int r = rowBase + rl;
#pragma unroll
    for (int kt = 0; kt < 4; ++kt)
      accum(afrag(A0, HIDDEN, r, kt * 32, hiHalf), W0, kt);
#pragma unroll
    for (int kt = 0; kt < 4; ++kt)
      accum(afrag(A1, HIDDEN, r, kt * 32, hiHalf), W0, kt + 4);
  }
  stash_relu(accs, b0, act, lane);

  // ---------------- layer 1 (LDS -> LDS) ----------------
  zero_accs();
#pragma unroll
  for (int kt = 0; kt < 4; ++kt)
    accum(afrag(act, HIDDEN, rl, kt * 32, hiHalf), W1, kt);
  stash_relu(accs, b1, act, lane);

  // ---------------- layer 2 (LDS -> global, no ReLU) ----------------
  {
    v8f acc2[N2T];
#pragma unroll
    for (int nt = 0; nt < N2T; ++nt) acc2[nt] = v8f{};
#pragma unroll
    for (int kt = 0; kt < 4; ++kt) {
      v16h a = afrag(act, HIDDEN, rl, kt * 32, hiHalf);
#pragma unroll
      for (int nt = 0; nt < N2T; ++nt) {
        v16h b = bfrag(W2, kt, nt, N2T, lane);
        acc2[nt] = WMMA_F16(a, b, acc2[nt]);
      }
    }
    const int r0 = (lane & 16) ? 8 : 0;
#pragma unroll
    for (int nt = 0; nt < N2T; ++nt) {
      const int col = nt * 16 + (lane & 15);
      const float bias = (col < n2cols) ? b2[col] : 0.0f;
#pragma unroll
      for (int r = 0; r < 8; ++r) {
        const int row = rowBase + r0 + r;
        if (row < M && col < n2cols)
          out[(size_t)row * ldo + col] = acc2[nt][r] + bias;
      }
    }
  }
}

// ---------------------------------------------------------------------------
// Wave32 reduction + LayerNorm epilogues (one wave per 128-wide row)
// ---------------------------------------------------------------------------
__device__ __forceinline__ float wave_sum(float v) {
#pragma unroll
  for (int m = 16; m >= 1; m >>= 1) v += __shfl_xor(v, m, 32);
  return v;
}

__global__ __launch_bounds__(256) void ln_post_kernel(
    const float* __restrict__ pre, int M, const float* __restrict__ g,
    const float* __restrict__ beta, const float* __restrict__ resid,
    float* __restrict__ dst32, _Float16* __restrict__ dst16) {
  const int row = blockIdx.x * 8 + (threadIdx.x >> 5);
  const int lane = threadIdx.x & 31;
  if (row >= M) return;
  const int c = lane * 4;
  const size_t base = (size_t)row * 128 + c;
  const float4 v = *(const float4*)(pre + base);
  const float mu = wave_sum(v.x + v.y + v.z + v.w) * (1.0f / 128.0f);
  const float dx = v.x - mu, dy = v.y - mu, dz = v.z - mu, dw = v.w - mu;
  const float var = wave_sum(dx * dx + dy * dy + dz * dz + dw * dw) * (1.0f / 128.0f);
  const float rstd = rsqrtf(var + 1e-5f);
  const float4 gg = *(const float4*)(g + c);
  const float4 bb = *(const float4*)(beta + c);
  float o0 = dx * rstd * gg.x + bb.x;
  float o1 = dy * rstd * gg.y + bb.y;
  float o2 = dz * rstd * gg.z + bb.z;
  float o3 = dw * rstd * gg.w + bb.w;
  if (resid) {
    const float4 rr = *(const float4*)(resid + base);
    o0 += rr.x; o1 += rr.y; o2 += rr.z; o3 += rr.w;
  }
  *(float4*)(dst32 + base) = make_float4(o0, o1, o2, o3);
  v4h h4 = {(_Float16)o0, (_Float16)o1, (_Float16)o2, (_Float16)o3};
  *(v4h*)(dst16 + base) = h4;
}

// Edge epilogue: LN + residual into e32/e16, fused scatter-sum into agg (f32 atomics).
__global__ __launch_bounds__(256) void edge_ln_scatter_kernel(
    const float* __restrict__ pre, int E_, const float* __restrict__ g,
    const float* __restrict__ beta, const int* __restrict__ dstIdx,
    float* __restrict__ e32, _Float16* __restrict__ e16,
    float* __restrict__ agg) {
  const int row = blockIdx.x * 8 + (threadIdx.x >> 5);
  const int lane = threadIdx.x & 31;
  if (row >= E_) return;
  const int c = lane * 4;
  const size_t base = (size_t)row * 128 + c;
  const float4 v = *(const float4*)(pre + base);
  const float mu = wave_sum(v.x + v.y + v.z + v.w) * (1.0f / 128.0f);
  const float dx = v.x - mu, dy = v.y - mu, dz = v.z - mu, dw = v.w - mu;
  const float var = wave_sum(dx * dx + dy * dy + dz * dz + dw * dw) * (1.0f / 128.0f);
  const float rstd = rsqrtf(var + 1e-5f);
  const float4 gg = *(const float4*)(g + c);
  const float4 bb = *(const float4*)(beta + c);
  const float4 old = *(const float4*)(e32 + base);
  const float o0 = dx * rstd * gg.x + bb.x + old.x;
  const float o1 = dy * rstd * gg.y + bb.y + old.y;
  const float o2 = dz * rstd * gg.z + bb.z + old.z;
  const float o3 = dw * rstd * gg.w + bb.w + old.w;
  *(float4*)(e32 + base) = make_float4(o0, o1, o2, o3);
  v4h h4 = {(_Float16)o0, (_Float16)o1, (_Float16)o2, (_Float16)o3};
  *(v4h*)(e16 + base) = h4;
  float* ap = agg + (size_t)dstIdx[row] * 128 + c;
  __hip_atomic_fetch_add(ap + 0, o0, __ATOMIC_RELAXED, __HIP_MEMORY_SCOPE_AGENT);
  __hip_atomic_fetch_add(ap + 1, o1, __ATOMIC_RELAXED, __HIP_MEMORY_SCOPE_AGENT);
  __hip_atomic_fetch_add(ap + 2, o2, __ATOMIC_RELAXED, __HIP_MEMORY_SCOPE_AGENT);
  __hip_atomic_fetch_add(ap + 3, o3, __ATOMIC_RELAXED, __HIP_MEMORY_SCOPE_AGENT);
}

// ---------------------------------------------------------------------------
// Small utility kernels
// ---------------------------------------------------------------------------
// fp32 weights [K x N] (optionally stacked over blockIdx.z) -> f16 B-fragments.
__global__ void frag_w_kernel(const float* __restrict__ W, int K, int N,
                              int ktiles, int ntiles, _Float16* __restrict__ dst) {
  const size_t per = (size_t)ktiles * ntiles * 512;
  const float* src = W + (size_t)blockIdx.z * K * N;
  _Float16* d = dst + (size_t)blockIdx.z * per;
  for (size_t t = (size_t)blockIdx.x * blockDim.x + threadIdx.x; t < per;
       t += (size_t)gridDim.x * blockDim.x) {
    const int tile = (int)(t >> 9);
    const int within = (int)(t & 511);
    const int lane = within >> 4;
    const int elem = within & 15;
    const int kt = tile / ntiles, nt = tile % ntiles;
    const int n = nt * 16 + (lane & 15);
    const int k = kt * 32 + ((lane & 16) ? 8 : 0) + (elem & 7) + ((elem & 8) ? 16 : 0);
    const float v = (k < K && n < N) ? src[(size_t)k * N + n] : 0.0f;
    d[t] = (_Float16)v;
  }
}

__global__ void convert_pad_kernel(const float* __restrict__ src, int rows, int cols,
                                   _Float16* __restrict__ dst, int drows, int dcols) {
  const size_t total = (size_t)drows * dcols;
  for (size_t t = (size_t)blockIdx.x * blockDim.x + threadIdx.x; t < total;
       t += (size_t)gridDim.x * blockDim.x) {
    const int r = (int)(t / dcols), c = (int)(t % dcols);
    const float v = (r < rows && c < cols) ? src[(size_t)r * cols + c] : 0.0f;
    dst[t] = (_Float16)v;
  }
}

__global__ void f32_to_f16_kernel(const float* __restrict__ src,
                                  _Float16* __restrict__ dst, size_t n) {
  for (size_t t = (size_t)blockIdx.x * blockDim.x + threadIdx.x; t < n;
       t += (size_t)gridDim.x * blockDim.x)
    dst[t] = (_Float16)src[t];
}

__global__ void zero_kernel(float* __restrict__ dst, size_t n) {
  for (size_t t = (size_t)blockIdx.x * blockDim.x + threadIdx.x; t < n;
       t += (size_t)gridDim.x * blockDim.x)
    dst[t] = 0.0f;
}

// ---------------------------------------------------------------------------
// Host orchestration
// ---------------------------------------------------------------------------
extern "C" void kernel_launch(void* const* d_in, const int* in_sizes, int n_in,
                              void* d_out, int out_size, void* d_ws, size_t ws_size,
                              hipStream_t stream) {
  (void)in_sizes; (void)out_size; (void)ws_size;
  constexpr int N = 10000, E = 80000, NPAD = 10112, MP = 15;
  if (n_in < 41) return;

  // ---- inputs (setup_inputs dict order; params flattened with sorted keys) ----
  int i = 0;
  const float* x  = (const float*)d_in[i++];   // [N,78]
  const float* ea = (const float*)d_in[i++];   // [E,3]
  // dec: b0,b1,b2,w0,w1,w2
  const float* dec_b0 = (const float*)d_in[i++];
  const float* dec_b1 = (const float*)d_in[i++];
  const float* dec_b2 = (const float*)d_in[i++];
  const float* dec_w0 = (const float*)d_in[i++];
  const float* dec_w1 = (const float*)d_in[i++];
  const float* dec_w2 = (const float*)d_in[i++];
  // edge_enc: b0,b1,b2,beta,g,w0,w1,w2
  const float* ee_b0 = (const float*)d_in[i++];
  const float* ee_b1 = (const float*)d_in[i++];
  const float* ee_b2 = (const float*)d_in[i++];
  const float* ee_be = (const float*)d_in[i++];
  const float* ee_g  = (const float*)d_in[i++];
  const float* ee_w0 = (const float*)d_in[i++];
  const float* ee_w1 = (const float*)d_in[i++];
  const float* ee_w2 = (const float*)d_in[i++];
  // node_enc
  const float* ne_b0 = (const float*)d_in[i++];
  const float* ne_b1 = (const float*)d_in[i++];
  const float* ne_b2 = (const float*)d_in[i++];
  const float* ne_be = (const float*)d_in[i++];
  const float* ne_g  = (const float*)d_in[i++];
  const float* ne_w0 = (const float*)d_in[i++];
  const float* ne_w1 = (const float*)d_in[i++];
  const float* ne_w2 = (const float*)d_in[i++];
  // proc.edge (stacked [15,...])
  const float* pe_b0 = (const float*)d_in[i++];
  const float* pe_b1 = (const float*)d_in[i++];
  const float* pe_b2 = (const float*)d_in[i++];
  const float* pe_be = (const float*)d_in[i++];
  const float* pe_g  = (const float*)d_in[i++];
  const float* pe_w0 = (const float*)d_in[i++];
  const float* pe_w1 = (const float*)d_in[i++];
  const float* pe_w2 = (const float*)d_in[i++];
  // proc.node (stacked [15,...])
  const float* pn_b0 = (const float*)d_in[i++];
  const float* pn_b1 = (const float*)d_in[i++];
  const float* pn_b2 = (const float*)d_in[i++];
  const float* pn_be = (const float*)d_in[i++];
  const float* pn_g  = (const float*)d_in[i++];
  const float* pn_w0 = (const float*)d_in[i++];
  const float* pn_w1 = (const float*)d_in[i++];
  const float* pn_w2 = (const float*)d_in[i++];
  const int* ei = (const int*)d_in[i++];       // [2,E]: src row, dst row

  // ---- workspace bump allocator ----
  char* wsBase = (char*)d_ws;
  size_t off = 0;
  auto alloc = [&](size_t bytes) -> void* {
    off = (off + 255) & ~(size_t)255;
    void* p = wsBase + off;
    off += bytes;
    return p;
  };
  _Float16* x16   = (_Float16*)alloc((size_t)NPAD * 96 * 2);
  _Float16* ea16  = (_Float16*)alloc((size_t)E * 32 * 2);
  _Float16* h16   = (_Float16*)alloc((size_t)NPAD * 128 * 2);
  _Float16* e16   = (_Float16*)alloc((size_t)E * 128 * 2);
  _Float16* agg16 = (_Float16*)alloc((size_t)NPAD * 128 * 2);
  float* h32   = (float*)alloc((size_t)NPAD * 128 * 4);
  float* e32   = (float*)alloc((size_t)E * 128 * 4);
  float* agg32 = (float*)alloc((size_t)NPAD * 128 * 4);
  float* tmp32 = (float*)alloc((size_t)E * 128 * 4);
  _Float16* wf_ne0 = (_Float16*)alloc((size_t)3 * 8 * 512 * 2);
  _Float16* wf_ne1 = (_Float16*)alloc((size_t)16384 * 2);
  _Float16* wf_ne2 = (_Float16*)alloc((size_t)16384 * 2);
  _Float16* wf_ee0 = (_Float16*)alloc((size_t)1 * 8 * 512 * 2);
  _Float16* wf_ee1 = (_Float16*)alloc((size_t)16384 * 2);
  _Float16* wf_ee2 = (_Float16*)alloc((size_t)16384 * 2);
  _Float16* wf_de0 = (_Float16*)alloc((size_t)16384 * 2);
  _Float16* wf_de1 = (_Float16*)alloc((size_t)16384 * 2);
  _Float16* wf_de2 = (_Float16*)alloc((size_t)4 * 5 * 512 * 2);
  _Float16* wf_pe0 = (_Float16*)alloc((size_t)MP * 12 * 8 * 512 * 2);
  _Float16* wf_pe1 = (_Float16*)alloc((size_t)MP * 16384 * 2);
  _Float16* wf_pe2 = (_Float16*)alloc((size_t)MP * 16384 * 2);
  _Float16* wf_pn0 = (_Float16*)alloc((size_t)MP * 8 * 8 * 512 * 2);
  _Float16* wf_pn1 = (_Float16*)alloc((size_t)MP * 16384 * 2);
  _Float16* wf_pn2 = (_Float16*)alloc((size_t)MP * 16384 * 2);

  // ---- weight fragmentization (fp32 -> f16 WMMA B fragments) ----
  auto frag = [&](const float* W, int K, int Nc, int kt, int nt, _Float16* dst, int z) {
    const size_t per = (size_t)kt * nt * 512;
    const int blocks = (int)((per + 255) / 256);
    frag_w_kernel<<<dim3(blocks, 1, z), dim3(256, 1, 1), 0, stream>>>(W, K, Nc, kt, nt, dst);
  };
  frag(ne_w0, 78, 128, 3, 8, wf_ne0, 1);
  frag(ne_w1, 128, 128, 4, 8, wf_ne1, 1);
  frag(ne_w2, 128, 128, 4, 8, wf_ne2, 1);
  frag(ee_w0, 3, 128, 1, 8, wf_ee0, 1);
  frag(ee_w1, 128, 128, 4, 8, wf_ee1, 1);
  frag(ee_w2, 128, 128, 4, 8, wf_ee2, 1);
  frag(dec_w0, 128, 128, 4, 8, wf_de0, 1);
  frag(dec_w1, 128, 128, 4, 8, wf_de1, 1);
  frag(dec_w2, 128, 78, 4, 5, wf_de2, 1);
  frag(pe_w0, 384, 128, 12, 8, wf_pe0, MP);
  frag(pe_w1, 128, 128, 4, 8, wf_pe1, MP);
  frag(pe_w2, 128, 128, 4, 8, wf_pe2, MP);
  frag(pn_w0, 256, 128, 8, 8, wf_pn0, MP);
  frag(pn_w1, 128, 128, 4, 8, wf_pn1, MP);
  frag(pn_w2, 128, 128, 4, 8, wf_pn2, MP);

  // ---- input conversion (fp32 -> f16, K padded to tile multiples) ----
  convert_pad_kernel<<<(NPAD * 96 + 255) / 256, 256, 0, stream>>>(x, N, 78, x16, NPAD, 96);
  convert_pad_kernel<<<(E * 32 + 255) / 256, 256, 0, stream>>>(ea, E, 3, ea16, E, 32);

  // ---- encoders ----
  mlp3_kernel<0, 3, 8><<<NPAD / 128, 256, 0, stream>>>(x16, 96, nullptr, nullptr, nullptr,
      N, wf_ne0, ne_b0, wf_ne1, ne_b1, wf_ne2, ne_b2, 128, tmp32, 128);
  ln_post_kernel<<<(N + 7) / 8, 256, 0, stream>>>(tmp32, N, ne_g, ne_be, nullptr, h32, h16);

  mlp3_kernel<0, 1, 8><<<E / 128, 256, 0, stream>>>(ea16, 32, nullptr, nullptr, nullptr,
      E, wf_ee0, ee_b0, wf_ee1, ee_b1, wf_ee2, ee_b2, 128, tmp32, 128);
  ln_post_kernel<<<(E + 7) / 8, 256, 0, stream>>>(tmp32, E, ee_g, ee_be, nullptr, e32, e16);

  // ---- message-passing processor ----
  for (int it = 0; it < MP; ++it) {
    const _Float16* W0 = wf_pe0 + (size_t)it * 12 * 8 * 512;
    const _Float16* W1 = wf_pe1 + (size_t)it * 16384;
    const _Float16* W2 = wf_pe2 + (size_t)it * 16384;
    // Edge MLP on concat(h[src], h[dst], e) gathered in-kernel.
    mlp3_kernel<1, 12, 8><<<E / 128, 256, 0, stream>>>(h16, HIDDEN, e16, ei, ei + E,
        E, W0, pe_b0 + it * 128, W1, pe_b1 + it * 128, W2, pe_b2 + it * 128,
        128, tmp32, 128);
    zero_kernel<<<(NPAD * 128 + 255) / 256, 256, 0, stream>>>(agg32, (size_t)NPAD * 128);
    edge_ln_scatter_kernel<<<(E + 7) / 8, 256, 0, stream>>>(
        tmp32, E, pe_g + it * 128, pe_be + it * 128, ei + E, e32, e16, agg32);
    f32_to_f16_kernel<<<(NPAD * 128 + 255) / 256, 256, 0, stream>>>(agg32, agg16,
                                                                    (size_t)NPAD * 128);
    // Node MLP on concat(h, agg).
    mlp3_kernel<2, 8, 8><<<NPAD / 128, 256, 0, stream>>>(h16, HIDDEN, agg16, nullptr,
        nullptr, N, wf_pn0 + (size_t)it * 8 * 8 * 512, pn_b0 + it * 128,
        wf_pn1 + (size_t)it * 16384, pn_b1 + it * 128,
        wf_pn2 + (size_t)it * 16384, pn_b2 + it * 128, 128, tmp32, 128);
    ln_post_kernel<<<(N + 7) / 8, 256, 0, stream>>>(tmp32, N, pn_g + it * 128,
                                                    pn_be + it * 128, h32, h32, h16);
  }

  // ---- decoder: h -> [N,78] straight into d_out ----
  mlp3_kernel<0, 4, 5><<<NPAD / 128, 256, 0, stream>>>(h16, HIDDEN, nullptr, nullptr,
      nullptr, N, wf_de0, dec_b0, wf_de1, dec_b1, wf_de2, dec_b2, 78,
      (float*)d_out, 78);
}
